// TFN_76716705841763
// MI455X (gfx1250) — matrix operand order repaired and checked by reading, not compiled, not run on hardware
//
#include <hip/hip_runtime.h>
#include <hip/hip_bf16.h>

typedef _Float16    v16h __attribute__((ext_vector_type(16)));
typedef float       v8f  __attribute__((ext_vector_type(8)));
typedef unsigned int v4u __attribute__((ext_vector_type(4)));
typedef int         v8i  __attribute__((ext_vector_type(8)));
typedef int         v4i  __attribute__((ext_vector_type(4)));

#define NPAIR  8385      // 129*65 (i,j) pairs
#define PPC    66        // pairs per K-chunk
#define NCHUNK 128       // ceil(8385/66)
#define POST   128
#define LDSTR  130       // padded LDS row stride (floats): 16*130 % 64 == 32 -> no bank conflict

// ---------------------------------------------------------------------------
// TDM: load 64 contiguous W1 rows (rows pair*65+1 .. pair*65+64, 128 f32 each)
// into LDS at byte offset lds_off, inserting 2 DWORDs of padding after every
// 128 DWORDs (row stride 130 floats).  D# per CDNA5 ISA §8.3/8.4.
// ---------------------------------------------------------------------------
__device__ __forceinline__ void tdm_load_wchunk(const float* W1, int pair, unsigned lds_off) {
    unsigned long long ga = (unsigned long long)(uintptr_t)W1
                          + ((unsigned long long)pair * 65ull + 1ull) * 128ull * 4ull;
    v4u g0;
    g0[0] = 1u;                                   // count=1 (valid), user mode
    g0[1] = lds_off;                              // lds_addr (bytes)
    g0[2] = (unsigned)(ga & 0xFFFFFFFFull);       // global_addr[31:0]
    g0[3] = (unsigned)((ga >> 32) & 0x1FFFFFFu)   // global_addr[56:32]
          | (2u << 30);                           // type = 2 ("image")
    v8i g1;
    g1[0] = (int)((2u << 16)                      // data_size = 2 (4B)
                | (1u << 20)                      // pad_enable
                | (6u << 22)                      // pad_interval: 128 DWORDs
                | (1u << 25));                    // pad_amount:   2 DWORDs
    g1[1] = (int)(128u << 16);                    // tensor_dim0 = 128
    g1[2] = (int)(64u  << 16);                    // tensor_dim1 = 64
    g1[3] = (int)(128u << 16);                    // tile_dim0 = 128
    g1[4] = (int)(64u);                           // tile_dim1 = 64, tile_dim2 = 0
    g1[5] = (int)(128u);                          // tensor_dim0_stride = 128
    g1[6] = 0;                                    // dim1 stride unused (2D tile)
    g1[7] = 0;
    v4i z4 = {0, 0, 0, 0};                        // groups 2/3 unused (2D)
    v8i z8 = {0, 0, 0, 0, 0, 0, 0, 0};
    __builtin_amdgcn_tensor_load_to_lds(g0, g1, z4, z4, z8, 0);
}

// ---------------------------------------------------------------------------
// Kernel 1: zero the Y1 accumulator [512 x 128] f32 in workspace
// ---------------------------------------------------------------------------
__global__ void tfn_init(float* __restrict__ Y1ws) {
    int idx = blockIdx.x * blockDim.x + threadIdx.x;
    if (idx < 512 * POST) Y1ws[idx] = 0.0f;
}

// ---------------------------------------------------------------------------
// Kernel 2: Y1ws[b,p] += sum over pair-chunk of s(b,pair) *
//              ( W1[pair*65, p] + sum_k a[b,k] * W1[pair*65+1+k, p] )
// grid = (4 M-blocks, NCHUNK K-chunks), block = 256 (8 waves as 4M x 2N grid,
// each wave: 2 M-subtiles x 4 N-subtiles of 16x16 WMMA tiles)
// ---------------------------------------------------------------------------
__global__ __launch_bounds__(256) void tfn_main(
    const float* __restrict__ audio,   // [512, 64]
    const float* __restrict__ video,   // [512, 64]
    const float* __restrict__ text,    // [512, 128]
    const float* __restrict__ W1,      // [545025, 128]
    float*       __restrict__ Y1ws)    // [512, 128] accumulator
{
    __shared__ float ldsW[2][64 * LDSTR];         // 2 x 32.5KB double buffer

    const int wid  = threadIdx.x >> 5;            // wave id 0..7
    const int lane = threadIdx.x & 31;
    const int ln   = lane & 15;                   // lane % 16
    const int hi   = lane >> 4;                   // lane / 16
    const int mg   = wid >> 1;                    // M-group 0..3  (32 rows each)
    const int ng   = wid & 1;                     // N-group 0..1  (64 cols each)
    const int base = blockIdx.x * 128 + mg * 32;  // wave's batch-row base

    // --- A fragments per M-subtile (16x32 f16 layout):
    //     kstep0 = a[0..31], kstep1 = a[32..63]
    v16h a0[2], a1[2];
    #pragma unroll
    for (int m = 0; m < 2; ++m) {
        const float* ar = audio + (size_t)(base + 16 * m + ln) * 64 + 8 * hi;
        #pragma unroll
        for (int e = 0; e < 8; ++e) {
            a0[m][e]     = (_Float16)ar[e];
            a0[m][8 + e] = (_Float16)ar[16 + e];
            a1[m][e]     = (_Float16)ar[32 + e];
            a1[m][8 + e] = (_Float16)ar[48 + e];
        }
    }

    v8f acc[2][4];
    #pragma unroll
    for (int m = 0; m < 2; ++m)
        #pragma unroll
        for (int n = 0; n < 4; ++n)
            #pragma unroll
            for (int r = 0; r < 8; ++r) acc[m][n][r] = 0.0f;

    const int p0   = blockIdx.y * PPC;
    const int pend = (p0 + PPC < NPAIR) ? (p0 + PPC) : NPAIR;
    int i = p0 / 65, j = p0 % 65;

    const unsigned ldsoff0 = (unsigned)(uintptr_t)(void*)&ldsW[0][0];
    const unsigned ldsoff1 = (unsigned)(uintptr_t)(void*)&ldsW[1][0];

    if (wid == 0) tdm_load_wchunk(W1, p0, ldsoff0);   // prologue

    for (int p = p0; p < pend; ++p) {
        const int    cur = (p - p0) & 1;
        const float* wb  = ldsW[cur];

        if (wid == 0) {
            if (p + 1 < pend) {
                tdm_load_wchunk(W1, p + 1, cur ? ldsoff0 : ldsoff1);
                __builtin_amdgcn_s_wait_tensorcnt(1);   // current buffer done
            } else {
                __builtin_amdgcn_s_wait_tensorcnt(0);
            }
        }
        __syncthreads();   // buffer `cur` ready for all waves

        // per-row scale s(b) = t1[b,i] * v1[b,j]  (aug: index 0 -> 1.0)
        float sv[2];
        v16h  sa0[2], sa1[2];
        #pragma unroll
        for (int m = 0; m < 2; ++m) {
            const int row = base + 16 * m + ln;
            float tv = (i == 0) ? 1.0f : text [(size_t)row * 128 + (i - 1)];
            float vv = (j == 0) ? 1.0f : video[(size_t)row * 64  + (j - 1)];
            sv[m] = tv * vv;
            _Float16 sh = (_Float16)sv[m];
            sa0[m] = a0[m] * sh;            // fold scale into A (v_pk_mul_f16)
            sa1[m] = a1[m] * sh;
        }
        // cross-lane s for the C-layout rows (M = r + 8*hi)
        float sr[2][8];
        #pragma unroll
        for (int m = 0; m < 2; ++m)
            #pragma unroll
            for (int r = 0; r < 8; ++r) sr[m][r] = __shfl(sv[m], 8 * hi + r, 32);

        const float* w0row = W1 + (size_t)p * 65 * 128;   // L2/L0-cached broadcast

        #pragma unroll
        for (int n = 0; n < 4; ++n) {
            const int    nc  = 16 * (4 * ng + n);
            const float* col = wb + nc + ln;
            v16h bf0, bf1;
            #pragma unroll
            for (int e = 0; e < 16; ++e) {
                bf0[e] = (_Float16)col[(     8 * hi + (e & 7) + 16 * (e >> 3)) * LDSTR];  // rows 0..31
                bf1[e] = (_Float16)col[(32 + 8 * hi + (e & 7) + 16 * (e >> 3)) * LDSTR];  // rows 32..63
            }
            // two independent accumulator chains interleaved per kstep
            acc[0][n] = __builtin_amdgcn_wmma_f32_16x16x32_f16(
                            false, sa0[0], false, bf0, (short)0, acc[0][n], false, false);
            acc[1][n] = __builtin_amdgcn_wmma_f32_16x16x32_f16(
                            false, sa0[1], false, bf0, (short)0, acc[1][n], false, false);
            acc[0][n] = __builtin_amdgcn_wmma_f32_16x16x32_f16(
                            false, sa1[0], false, bf1, (short)0, acc[0][n], false, false);
            acc[1][n] = __builtin_amdgcn_wmma_f32_16x16x32_f16(
                            false, sa1[1], false, bf1, (short)0, acc[1][n], false, false);

            // augmented "1" element of a1aug: Y1 += s(b) * W1[pair*65, p]
            float w0 = w0row[nc + ln];
            #pragma unroll
            for (int m = 0; m < 2; ++m)
                #pragma unroll
                for (int r = 0; r < 8; ++r) acc[m][n][r] += sr[m][r] * w0;
        }

        if (++j == 65) { j = 0; ++i; }
        __syncthreads();   // all waves done reading buffer `cur`
    }

    // flush partial tiles: element (M = r + 8*hi, N = ln) of each subtile
    #pragma unroll
    for (int m = 0; m < 2; ++m)
        #pragma unroll
        for (int n = 0; n < 4; ++n)
            #pragma unroll
            for (int r = 0; r < 8; ++r)
                atomicAdd(&Y1ws[(size_t)(base + 16 * m + 8 * hi + r) * POST
                                + 16 * (4 * ng + n) + ln],
                          acc[m][n][r]);
}

// ---------------------------------------------------------------------------
// Kernel 3: MLP head.  block = 128 threads handles one batch row.
// ---------------------------------------------------------------------------
__global__ __launch_bounds__(128) void tfn_mlp(
    const float* __restrict__ Y1ws, const float* __restrict__ b1,
    const float* __restrict__ W2,   const float* __restrict__ b2,
    const float* __restrict__ W3,   const float* __restrict__ b3,
    float* __restrict__ out)
{
    __shared__ float y1s[POST];
    __shared__ float red[POST];
    const int p = threadIdx.x;
    const int b = blockIdx.x;

    float y1 = Y1ws[(size_t)b * POST + p] + b1[p];
    y1s[p] = fmaxf(y1, 0.0f);
    __syncthreads();

    float acc = b2[p];
    #pragma unroll 4
    for (int q = 0; q < POST; ++q) acc += y1s[q] * W2[q * POST + p];
    float y2 = fmaxf(acc, 0.0f);

    red[p] = y2 * W3[p];
    __syncthreads();
    #pragma unroll
    for (int s = 64; s > 0; s >>= 1) {
        if (p < s) red[p] += red[p + s];
        __syncthreads();
    }
    if (p == 0) {
        float z  = red[0] + b3[0];
        float sg = 1.0f / (1.0f + __expf(-z));
        out[b] = sg * 6.0f - 3.0f;
    }
}

// ---------------------------------------------------------------------------
extern "C" void kernel_launch(void* const* d_in, const int* in_sizes, int n_in,
                              void* d_out, int out_size, void* d_ws, size_t ws_size,
                              hipStream_t stream) {
    const float* audio = (const float*)d_in[0];
    const float* video = (const float*)d_in[1];
    const float* text  = (const float*)d_in[2];
    const float* W1    = (const float*)d_in[3];
    const float* b1    = (const float*)d_in[4];
    const float* W2    = (const float*)d_in[5];
    const float* b2    = (const float*)d_in[6];
    const float* W3    = (const float*)d_in[7];
    const float* b3    = (const float*)d_in[8];

    float* Y1ws = (float*)d_ws;            // 512*128*4 = 256 KB scratch
    float* outp = (float*)d_out;           // [512] f32

    tfn_init<<<256, 256, 0, stream>>>(Y1ws);
    tfn_main<<<dim3(4, NCHUNK), 256, 0, stream>>>(audio, video, text, W1, Y1ws);
    tfn_mlp<<<512, 128, 0, stream>>>(Y1ws, b1, W2, b2, W3, b3, outp);
}
